// FusedKLDivLoss_14577119003196
// MI455X (gfx1250) — matrix operand order repaired and checked by reading, not compiled, not run on hardware
//
#include <hip/hip_runtime.h>
#include <hip/hip_bf16.h>

// ---------------------------------------------------------------------------
// Fused KD loss: KL(softmax(target_x @ target_Wt) || softmax(x @ Wt)) / N
//   N=2048, H=4096, V=32000.  bf16 WMMA GEMM + flash-style online softmax;
//   logits never materialized. Weight stream is the roofline term, so each
//   wave now computes TWO 16-row M-tiles per B operand (32-row blocks) to
//   double MACs per weight byte. x tiles staged via async global->LDS loads.
// ---------------------------------------------------------------------------

typedef __bf16 bf16_t;
typedef __attribute__((ext_vector_type(16))) __bf16 v16bf;
typedef __attribute__((ext_vector_type(8)))  __bf16 v8bf;
typedef __attribute__((ext_vector_type(8)))  float  v8f;

#define N_ROWS 2048
#define H_DIM  4096
#define V_DIM  32000
#define SPLITS 5
#define V_PER_SPLIT (V_DIM / SPLITS)            // 6400
#define TILE_V 128                              // 8 waves x 16 cols
#define TILES_PER_SPLIT (V_PER_SPLIT / TILE_V)  // 50
#define K_CHUNK 256
#define ROW_BLOCK 32                            // two 16-row WMMA tiles
#define LDS_PAD 8                               // 16B pad -> conflict-free b128
#define LDS_STRIDE (K_CHUNK + LDS_PAD)

__device__ __forceinline__ v16bf pack16(uint4 lo, uint4 hi) {
  union { uint4 q[2]; v16bf v; } u;
  u.q[0] = lo; u.q[1] = hi;
  return u.v;
}

// async global -> LDS copy of 16 bytes per lane (gfx1250 ASYNCcnt path)
__device__ __forceinline__ void async_copy_b128(void* lds_dst, const void* gsrc) {
  unsigned lds_off = (unsigned)(size_t)lds_dst;        // low 32 bits = LDS addr
  unsigned long long gaddr = (unsigned long long)(size_t)gsrc;
  asm volatile("global_load_async_to_lds_b128 %0, %1, off"
               :
               : "v"(lds_off), "v"(gaddr)
               : "memory");
}
__device__ __forceinline__ void wait_asynccnt0() {
  asm volatile("s_wait_asynccnt 0x0" ::: "memory");
}

// ---------------------------------------------------------------------------
// fp32 -> bf16 bulk convert (n multiple of 8)
// ---------------------------------------------------------------------------
__global__ void cvt_f32_bf16(const float* __restrict__ src,
                             bf16_t* __restrict__ dst, size_t n) {
  size_t i = ((size_t)blockIdx.x * blockDim.x + threadIdx.x) * 8;
  size_t stride = (size_t)gridDim.x * blockDim.x * 8;
  for (; i < n; i += stride) {
    float4 a = *(const float4*)(src + i);
    float4 b = *(const float4*)(src + i + 4);
    v8bf o;
    o[0] = (bf16_t)a.x; o[1] = (bf16_t)a.y; o[2] = (bf16_t)a.z; o[3] = (bf16_t)a.w;
    o[4] = (bf16_t)b.x; o[5] = (bf16_t)b.y; o[6] = (bf16_t)b.z; o[7] = (bf16_t)b.w;
    *(v8bf*)(dst + i) = o;
  }
}

// ---------------------------------------------------------------------------
// Main fused kernel.
// grid = (N/32 row-blocks, SPLITS v-splits), block = 256 (8 wave32).
// Each wave: 16 columns, two 16-row M-tiles (j=0,1) for student & teacher.
// stats layout: [row][split][wave][5] = {m_t, Z_t, Sum p*(t-s), m_s, Z_s}
// ---------------------------------------------------------------------------
__global__ __launch_bounds__(256, 1)
void kd_main(const bf16_t* __restrict__ xs, const bf16_t* __restrict__ xt,
             const bf16_t* __restrict__ wsb, const bf16_t* __restrict__ wtb,
             float* __restrict__ stats) {
  __shared__ bf16_t lds_xs[ROW_BLOCK][LDS_STRIDE];
  __shared__ bf16_t lds_xt[ROW_BLOCK][LDS_STRIDE];

  const int tid   = threadIdx.x;
  const int wave  = tid >> 5;
  const int lane  = tid & 31;
  const int half  = lane >> 4;   // which 16-lane half
  const int lm    = lane & 15;
  const int n0    = blockIdx.x * ROW_BLOCK;
  const int split = blockIdx.y;

  // Online stats for 2 M-tiles x 8 rows (row = j*16 + half*8 + i)
  float mt[16], Zt[16], Pt[16], ms[16], Zs[16];
#pragma unroll
  for (int i = 0; i < 16; ++i) {
    mt[i] = -__builtin_inff(); ms[i] = -__builtin_inff();
    Zt[i] = 0.f; Zs[i] = 0.f; Pt[i] = 0.f;
  }

  for (int vt = 0; vt < TILES_PER_SPLIT; ++vt) {
    const int vcol = split * V_PER_SPLIT + vt * TILE_V + wave * 16 + lm;
    // B operand: lane = column v; K = {half*8 .. +7} and {half*8+16 .. +23}
    const bf16_t* bsp = wsb + (size_t)vcol * H_DIM + half * 8;
    const bf16_t* btp = wtb + (size_t)vcol * H_DIM + half * 8;

    v8f acc_s0 = {}, acc_s1 = {}, acc_t0 = {}, acc_t1 = {};

    for (int kc = 0; kc < H_DIM; kc += K_CHUNK) {
      __syncthreads();  // previous chunk fully consumed by all waves
      // async-stage 32 x 256 bf16 of x and target_x (16 KB each).
      // 2048 16-byte chunks total -> 8 per thread.
#pragma unroll
      for (int it = 0; it < 8; ++it) {
        int t = tid + it * 256;
        int m = t >> 10;            // 0: xs, 1: xt
        int r = (t >> 5) & 31;
        int c = (t & 31) * 8;
        const bf16_t* g = (m ? xt : xs) + (size_t)(n0 + r) * H_DIM + kc + c;
        void* l = m ? (void*)&lds_xt[r][c] : (void*)&lds_xs[r][c];
        async_copy_b128(l, g);
      }
      wait_asynccnt0();
      __syncthreads();

      // prefetch next weight chunk (streaming; global_prefetch_b8)
      if (kc + K_CHUNK < H_DIM) {
        __builtin_prefetch(bsp + kc + K_CHUNK, 0, 0);
        __builtin_prefetch(btp + kc + K_CHUNK, 0, 0);
        __builtin_prefetch(bsp + kc + K_CHUNK + 128, 0, 0);
        __builtin_prefetch(btp + kc + K_CHUNK + 128, 0, 0);
      }

#pragma unroll 2
      for (int kk = 0; kk < K_CHUNK; kk += 32) {
        // A operand (16-bit A 16x32 layout): row = lm (+16 for tile 1),
        // K = {half*8 .. +7} and {half*8+16 .. +23}
        const int ao = kk + half * 8;
        v16bf a_s0 = pack16(*(const uint4*)&lds_xs[lm][ao],
                            *(const uint4*)&lds_xs[lm][ao + 16]);
        v16bf a_s1 = pack16(*(const uint4*)&lds_xs[lm + 16][ao],
                            *(const uint4*)&lds_xs[lm + 16][ao + 16]);
        v16bf a_t0 = pack16(*(const uint4*)&lds_xt[lm][ao],
                            *(const uint4*)&lds_xt[lm][ao + 16]);
        v16bf a_t1 = pack16(*(const uint4*)&lds_xt[lm + 16][ao],
                            *(const uint4*)&lds_xt[lm + 16][ao + 16]);
        // B operand: two 8-K runs per lane, same K order as A (shared by both M-tiles)
        const int k = kc + kk;
        v16bf b_s = pack16(*(const uint4*)(bsp + k),
                           *(const uint4*)(bsp + k + 16));
        v16bf b_t = pack16(*(const uint4*)(btp + k),
                           *(const uint4*)(btp + k + 16));
        acc_s0 = __builtin_amdgcn_wmma_f32_16x16x32_bf16(
            false, a_s0, false, b_s, (short)0, acc_s0, false, false);
        acc_s1 = __builtin_amdgcn_wmma_f32_16x16x32_bf16(
            false, a_s1, false, b_s, (short)0, acc_s1, false, false);
        acc_t0 = __builtin_amdgcn_wmma_f32_16x16x32_bf16(
            false, a_t0, false, b_t, (short)0, acc_t0, false, false);
        acc_t1 = __builtin_amdgcn_wmma_f32_16x16x32_bf16(
            false, a_t1, false, b_t, (short)0, acc_t1, false, false);
      }
    }

    // Online update for both M-tiles; 16 tile columns live in a 16-lane half.
#pragma unroll
    for (int j = 0; j < 2; ++j) {
#pragma unroll
      for (int i = 0; i < 8; ++i) {
        float sv = j ? acc_s1[i] : acc_s0[i];
        float tv = j ? acc_t1[i] : acc_t0[i];
        const int s = j * 8 + i;
        // teacher-side
        float tmax = tv;
#pragma unroll
        for (int m = 1; m <= 8; m <<= 1) tmax = fmaxf(tmax, __shfl_xor(tmax, m, 32));
        float mtn = fmaxf(mt[s], tmax);
        float p  = __expf(tv - mtn);
        float pd = p * (tv - sv);
        float sp = p, spd = pd;
#pragma unroll
        for (int m = 1; m <= 8; m <<= 1) { sp += __shfl_xor(sp, m, 32); spd += __shfl_xor(spd, m, 32); }
        float sc = __expf(mt[s] - mtn);
        Zt[s] = Zt[s] * sc + sp;
        Pt[s] = Pt[s] * sc + spd;
        mt[s] = mtn;
        // student-side lse
        float smax = sv;
#pragma unroll
        for (int m = 1; m <= 8; m <<= 1) smax = fmaxf(smax, __shfl_xor(smax, m, 32));
        float msn = fmaxf(ms[s], smax);
        float q = __expf(sv - msn);
#pragma unroll
        for (int m = 1; m <= 8; m <<= 1) q += __shfl_xor(q, m, 32);
        Zs[s] = Zs[s] * __expf(ms[s] - msn) + q;
        ms[s] = msn;
      }
    }
  }

  // one lane per half writes its rows' partial stats (both M-tiles)
  if (lm == 0) {
#pragma unroll
    for (int j = 0; j < 2; ++j) {
#pragma unroll
      for (int i = 0; i < 8; ++i) {
        const int s = j * 8 + i;
        int row = n0 + j * 16 + half * 8 + i;
        size_t base = ((((size_t)row * SPLITS + split) * 8) + wave) * 5;
        stats[base + 0] = mt[s];
        stats[base + 1] = Zt[s];
        stats[base + 2] = Pt[s];
        stats[base + 3] = ms[s];
        stats[base + 4] = Zs[s];
      }
    }
  }
}

// ---------------------------------------------------------------------------
// Finalize: merge 5*8 partials per row (stable LSE merge), sum rows, /N.
// ---------------------------------------------------------------------------
__global__ void kd_finalize(const float* __restrict__ stats, float* __restrict__ out) {
  __shared__ float red[256];
  float local = 0.f;
  for (int row = threadIdx.x; row < N_ROWS; row += 256) {
    const float* p = stats + (size_t)row * SPLITS * 8 * 5;
    float mT = -__builtin_inff(), mS = -__builtin_inff();
    for (int j = 0; j < SPLITS * 8; ++j) {
      mT = fmaxf(mT, p[j * 5 + 0]);
      mS = fmaxf(mS, p[j * 5 + 3]);
    }
    float Zt = 0.f, Pt = 0.f, Zs = 0.f;
    for (int j = 0; j < SPLITS * 8; ++j) {
      float et = __expf(p[j * 5 + 0] - mT);
      Zt += p[j * 5 + 1] * et;
      Pt += p[j * 5 + 2] * et;
      Zs += p[j * 5 + 4] * __expf(p[j * 5 + 3] - mS);
    }
    // KL_row = E_t[t - s] - lse_t + lse_s
    local += Pt / Zt - (mT + __logf(Zt)) + (mS + __logf(Zs));
  }
  red[threadIdx.x] = local;
  __syncthreads();
  for (int s = 128; s > 0; s >>= 1) {
    if (threadIdx.x < s) red[threadIdx.x] += red[threadIdx.x + s];
    __syncthreads();
  }
  if (threadIdx.x == 0) out[0] = red[0] / (float)N_ROWS;
}

// ---------------------------------------------------------------------------
extern "C" void kernel_launch(void* const* d_in, const int* in_sizes, int n_in,
                              void* d_out, int out_size, void* d_ws, size_t ws_size,
                              hipStream_t stream) {
  const float* x  = (const float*)d_in[0];
  const float* tx = (const float*)d_in[1];
  const float* w  = (const float*)d_in[2];
  const float* tw = (const float*)d_in[3];
  float* out = (float*)d_out;

  const size_t nW = (size_t)V_DIM * H_DIM;
  const size_t nX = (size_t)N_ROWS * H_DIM;

  bf16_t* wbf_s = (bf16_t*)d_ws;
  bf16_t* wbf_t = wbf_s + nW;
  bf16_t* xbf_s = wbf_t + nW;
  bf16_t* xbf_t = xbf_s + nX;
  float*  stats = (float*)(xbf_t + nX);   // [N][SPLITS][8][5] floats

  cvt_f32_bf16<<<4096, 256, 0, stream>>>(w,  wbf_s, nW);
  cvt_f32_bf16<<<4096, 256, 0, stream>>>(tw, wbf_t, nW);
  cvt_f32_bf16<<<512,  256, 0, stream>>>(x,  xbf_s, nX);
  cvt_f32_bf16<<<512,  256, 0, stream>>>(tx, xbf_t, nX);

  dim3 grid(N_ROWS / ROW_BLOCK, SPLITS);
  kd_main<<<grid, 256, 0, stream>>>(xbf_s, xbf_t, wbf_s, wbf_t, stats);
  kd_finalize<<<1, 256, 0, stream>>>(stats, out);
}
